// GCN60_71511205478661
// MI455X (gfx1250) — matrix operand order repaired and checked by reading, not compiled, not run on hardware
//
#include <hip/hip_runtime.h>

// ---------------------------------------------------------------------------
// CDNA5 (gfx1250) GraphSAGE forward: 4x SAGEConv (WMMA bf16 GEMMs, LDS-staged
// weights) + per-graph mean pooling
// ---------------------------------------------------------------------------

typedef __attribute__((ext_vector_type(16))) __bf16 v16bf;
typedef __attribute__((ext_vector_type(8)))  float  v8f;

// ---------------- degree ----------------------------------------------------
__global__ void deg_count_kernel(const int* __restrict__ dst, float* __restrict__ deg, int E) {
    int e = blockIdx.x * blockDim.x + threadIdx.x;
    if (e < E) atomicAdd(&deg[dst[e]], 1.0f);
}

__global__ void inv_deg_kernel(float* __restrict__ deg, int N) {
    int n = blockIdx.x * blockDim.x + threadIdx.x;
    if (n < N) deg[n] = 1.0f / fmaxf(deg[n], 1.0f);
}

// ---------------- weight swizzle: fp32 [din,dout] -> bf16 WMMA B fragments --
// B (32x16, 16-bit) fragment layout: lane holds n = lane&15; element i holds
// k = kt*32 + 16*(lane>>4) + i. Stored so each lane reads 16 contiguous bf16.
__global__ void swizzle_weights_kernel(const float* __restrict__ W, __bf16* __restrict__ Wz,
                                       int din, int dout, int KT, int NT) {
    int idx = blockIdx.x * blockDim.x + threadIdx.x;
    int total = KT * NT * 32;
    if (idx >= total) return;
    int lane = idx & 31;
    int tile = idx >> 5;
    int nt   = tile % NT;
    int kt   = tile / NT;
    int n    = nt * 16 + (lane & 15);
    int half = lane >> 4;
    __bf16* out = Wz + (size_t)idx * 16;
#pragma unroll
    for (int i = 0; i < 16; ++i) {
        int k = kt * 32 + half * 16 + i;
        float v = (k < din && n < dout) ? W[(size_t)k * dout + n] : 0.0f;
        out[i] = (__bf16)v;
    }
}

// ---------------- edge scatter-add (mean numerator) -------------------------
template<int DIN>
__global__ void scatter_add_kernel(const float* __restrict__ h, const int* __restrict__ src,
                                   const int* __restrict__ dst, float* __restrict__ agg, int E) {
    int idx = blockIdx.x * blockDim.x + threadIdx.x;   // max E*DIN = 204.8M < 2^31
    int total = E * DIN;
    if (idx >= total) return;
    int e = idx / DIN;
    int f = idx - e * DIN;
    atomicAdd(&agg[(size_t)dst[e] * DIN + f], h[(size_t)src[e] * DIN + f]);
}

// ---------------- fused SAGE layer: out = relu(h@Ws + (agg/deg)@Wn + b) -----
// One wave per 16-row node tile. bf16 WMMA with f32 accumulation.
// Both swizzled weight matrices are staged into LDS once per workgroup so the
// WMMA inner loop feeds from ds_load_b128 instead of redundant L2 traffic.
template<int DIN, int DOUT, bool RELU>
__global__ void __launch_bounds__(256)
sage_wmma_kernel(const float* __restrict__ hin, const float* __restrict__ agg,
                 const float* __restrict__ inv_deg,
                 const __bf16* __restrict__ Wsz, const __bf16* __restrict__ Wnz,
                 const float* __restrict__ bias, float* __restrict__ out,
                 int N, int n_tiles) {
    constexpr int KT   = (DIN + 31) / 32;
    constexpr int NT   = (DOUT + 15) / 16;
    constexpr int FRAG = KT * NT * 32 * 16;        // bf16 elems per weight matrix

    __shared__ __align__(32) __bf16 ldsW[2 * FRAG];

    // ---- cooperative stage of both weight matrices into LDS (128-bit copies)
    {
        constexpr int VECS = (FRAG * 2) / 16;      // uint4 count per matrix
        uint4* l = (uint4*)ldsW;
        const uint4* g0 = (const uint4*)Wsz;
        const uint4* g1 = (const uint4*)Wnz;
        for (int i = threadIdx.x; i < VECS; i += 256) l[i] = g0[i];
        for (int i = threadIdx.x; i < VECS; i += 256) l[VECS + i] = g1[i];
    }
    __syncthreads();                               // all 8 waves participate

    const int wave = blockIdx.x * (blockDim.x >> 5) + (threadIdx.x >> 5);
    if (wave >= n_tiles) return;                   // wave-uniform: EXEC stays all-1s
    const int lane = threadIdx.x & 31;
    const int half = lane >> 4;
    const int col  = lane & 15;

    int m = wave * 16 + col;                       // A row this lane supplies
    if (m >= N) m = N - 1;
    const float s = inv_deg[m];
    const float* rowS = hin + (size_t)m * DIN;
    const float* rowN = agg + (size_t)m * DIN;

    v8f acc[NT];
#pragma unroll
    for (int nt = 0; nt < NT; ++nt) {
        const int n = nt * 16 + col;
        const float bv = (n < DOUT) ? bias[n] : 0.0f;
#pragma unroll
        for (int v = 0; v < 8; ++v) acc[nt][v] = bv;
    }

#pragma unroll
    for (int kt = 0; kt < KT; ++kt) {
        // A (16x32, 16-bit) layout: element i -> k = 16*half+i (matches B fragments)
        v16bf a_s, a_n;
#pragma unroll
        for (int i = 0; i < 8; ++i) {
            const int ka = kt * 32 + half * 16 + i;
            const int kb = kt * 32 + half * 16 + 8 + i;
            const float vs0 = (ka < DIN) ? rowS[ka]     : 0.0f;
            const float vn0 = (ka < DIN) ? rowN[ka] * s : 0.0f;
            const float vs1 = (kb < DIN) ? rowS[kb]     : 0.0f;
            const float vn1 = (kb < DIN) ? rowN[kb] * s : 0.0f;
            a_s[i]     = (__bf16)vs0;
            a_s[i + 8] = (__bf16)vs1;
            a_n[i]     = (__bf16)vn0;
            a_n[i + 8] = (__bf16)vn1;
        }
#pragma unroll
        for (int nt = 0; nt < NT; ++nt) {
            const v16bf bs = *(const v16bf*)&ldsW[((kt * NT + nt) * 32 + lane) * 16];
            const v16bf bn = *(const v16bf*)&ldsW[FRAG + ((kt * NT + nt) * 32 + lane) * 16];
            acc[nt] = __builtin_amdgcn_wmma_f32_16x16x32_bf16(
                false, a_s, false, bs, (short)0, acc[nt], false, false);
            acc[nt] = __builtin_amdgcn_wmma_f32_16x16x32_bf16(
                false, a_n, false, bn, (short)0, acc[nt], false, false);
        }
    }

    // D layout: VGPR v, row = v + 8*half, col = lane&15
#pragma unroll
    for (int nt = 0; nt < NT; ++nt) {
        const int n = nt * 16 + col;
        if (n < DOUT) {
#pragma unroll
            for (int v = 0; v < 8; ++v) {
                const int row = wave * 16 + v + 8 * half;
                if (row < N) {
                    float val = acc[nt][v];
                    if (RELU) val = fmaxf(val, 0.0f);
                    out[(size_t)row * DOUT + n] = val;
                }
            }
        }
    }
}

// ---------------- per-graph mean pooling ------------------------------------
__global__ void pool_scatter_kernel(const float* __restrict__ h, const int* __restrict__ gids,
                                    float* __restrict__ psum, float* __restrict__ pcnt,
                                    int N, int C) {
    int n = blockIdx.x * blockDim.x + threadIdx.x;
    if (n >= N) return;
    int g = gids[n];
    atomicAdd(&pcnt[g], 1.0f);
    for (int f = 0; f < C; ++f)
        atomicAdd(&psum[g * C + f], h[(size_t)n * C + f]);
}

__global__ void pool_finalize_kernel(const float* __restrict__ psum, const float* __restrict__ pcnt,
                                     float* __restrict__ out, int total, int C) {
    int i = blockIdx.x * blockDim.x + threadIdx.x;
    if (i < total) out[i] = psum[i] / pcnt[i / C];
}

// ---------------------------------------------------------------------------
extern "C" void kernel_launch(void* const* d_in, const int* in_sizes, int n_in,
                              void* d_out, int out_size, void* d_ws, size_t ws_size,
                              hipStream_t stream) {
    const float* in_feat = (const float*)d_in[0];
    const int*   src     = (const int*)d_in[1];
    const int*   dst     = (const int*)d_in[2];
    const int*   gids    = (const int*)d_in[3];
    const float* W[8];
    const float* bias[4];
    for (int l = 0; l < 4; ++l) {
        W[2 * l]     = (const float*)d_in[4 + 3 * l];  // ws
        W[2 * l + 1] = (const float*)d_in[5 + 3 * l];  // wn
        bias[l]      = (const float*)d_in[6 + 3 * l];
    }
    const int N  = in_sizes[3];    // graph_ids length == N_NODES
    const int E  = in_sizes[1];    // edges
    const int NG = out_size / 5;

    static const int DINS[4]  = {128, 128, 118, 103};
    static const int DOUTS[4] = {128, 118, 103, 5};

    // ---- carve workspace (256-B aligned slices) ----
    size_t off = 0;
    auto carve = [&](size_t bytes) -> char* {
        char* p = (char*)d_ws + off;
        off += (bytes + 255) & ~(size_t)255;
        return p;
    };
    float* deg  = (float*)carve((size_t)N * 4);          // then holds 1/clamp(deg,1)
    float* bufA = (float*)carve((size_t)N * 128 * 4);
    float* bufB = (float*)carve((size_t)N * 128 * 4);
    float* agg  = (float*)carve((size_t)N * 128 * 4);
    __bf16* Wz[8];
    int KT[4], NT[4];
    for (int l = 0; l < 4; ++l) {
        KT[l] = (DINS[l] + 31) / 32;
        NT[l] = (DOUTS[l] + 15) / 16;
        size_t sz = (size_t)KT[l] * NT[l] * 32 * 16 * sizeof(__bf16);
        Wz[2 * l]     = (__bf16*)carve(sz);
        Wz[2 * l + 1] = (__bf16*)carve(sz);
    }
    float* psum = (float*)carve((size_t)out_size * 4);
    float* pcnt = (float*)carve((size_t)NG * 4);
    (void)ws_size;

    // ---- degrees ----
    hipMemsetAsync(deg, 0, (size_t)N * 4, stream);
    deg_count_kernel<<<(E + 255) / 256, 256, 0, stream>>>(dst, deg, E);
    inv_deg_kernel<<<(N + 255) / 256, 256, 0, stream>>>(deg, N);

    // ---- swizzle weights to bf16 WMMA fragments ----
    for (int l = 0; l < 4; ++l) {
        int total = KT[l] * NT[l] * 32;
        int blk = (total + 255) / 256;
        swizzle_weights_kernel<<<blk, 256, 0, stream>>>(W[2 * l],     Wz[2 * l],
                                                        DINS[l], DOUTS[l], KT[l], NT[l]);
        swizzle_weights_kernel<<<blk, 256, 0, stream>>>(W[2 * l + 1], Wz[2 * l + 1],
                                                        DINS[l], DOUTS[l], KT[l], NT[l]);
    }

    const int n_tiles = (N + 15) / 16;
    const int gemm_blocks = (n_tiles + 7) / 8;   // 256 thr = 8 waves/block

    // ---- layer 1: in_feat -> bufA ----
    hipMemsetAsync(agg, 0, (size_t)N * 128 * 4, stream);
    scatter_add_kernel<128><<<(E * 128 + 255) / 256, 256, 0, stream>>>(in_feat, src, dst, agg, E);
    sage_wmma_kernel<128, 128, true><<<gemm_blocks, 256, 0, stream>>>(
        in_feat, agg, deg, Wz[0], Wz[1], bias[0], bufA, N, n_tiles);

    // ---- layer 2: bufA -> bufB ----
    hipMemsetAsync(agg, 0, (size_t)N * 128 * 4, stream);
    scatter_add_kernel<128><<<(E * 128 + 255) / 256, 256, 0, stream>>>(bufA, src, dst, agg, E);
    sage_wmma_kernel<128, 118, true><<<gemm_blocks, 256, 0, stream>>>(
        bufA, agg, deg, Wz[2], Wz[3], bias[1], bufB, N, n_tiles);

    // ---- layer 3: bufB -> bufA ----
    hipMemsetAsync(agg, 0, (size_t)N * 118 * 4, stream);
    scatter_add_kernel<118><<<(E * 118 + 255) / 256, 256, 0, stream>>>(bufB, src, dst, agg, E);
    sage_wmma_kernel<118, 103, true><<<gemm_blocks, 256, 0, stream>>>(
        bufB, agg, deg, Wz[4], Wz[5], bias[2], bufA, N, n_tiles);

    // ---- layer 4 (no relu): bufA -> bufB ----
    hipMemsetAsync(agg, 0, (size_t)N * 103 * 4, stream);
    scatter_add_kernel<103><<<(E * 103 + 255) / 256, 256, 0, stream>>>(bufA, src, dst, agg, E);
    sage_wmma_kernel<103, 5, false><<<gemm_blocks, 256, 0, stream>>>(
        bufA, agg, deg, Wz[6], Wz[7], bias[3], bufB, N, n_tiles);

    // ---- per-graph mean pooling -> d_out ----
    hipMemsetAsync(psum, 0, (size_t)out_size * 4, stream);
    hipMemsetAsync(pcnt, 0, (size_t)NG * 4, stream);
    pool_scatter_kernel<<<(N + 255) / 256, 256, 0, stream>>>(bufB, gids, psum, pcnt, N, 5);
    pool_finalize_kernel<<<(out_size + 255) / 256, 256, 0, stream>>>(
        psum, pcnt, (float*)d_out, out_size, 5);
}
